// MultiBankIterativeBlock_76579266888374
// MI455X (gfx1250) — compile-verified, matrix operand order checked by reading, not run on hardware
//
#include <hip/hip_runtime.h>
#include <hip/hip_bf16.h>
#include <math.h>

typedef __attribute__((ext_vector_type(16))) _Float16 v16h;
typedef __attribute__((ext_vector_type(8)))  _Float16 v8h;
typedef __attribute__((ext_vector_type(8)))  float    v8f;

#define PI_F 3.14159265358979323846f

// ---------------------------------------------------------------------------
// WMMA helpers (CDNA5 v_wmma_f32_16x16x32_f16, wave32).
// 16-bit A (16x32) / B (32x16) operand striping per ISA 7.12.2:
//   lane = idx + 16*h  (idx = M for A, N for B), the 16 halves of the lane
//   cover kk = {8h..8h+7} and {16+8h..16+8h+7}  -> two 16-byte loads.
// ---------------------------------------------------------------------------
__device__ __forceinline__ v16h load_op32(const _Float16* p, int h) {
  const v8h lo = *reinterpret_cast<const v8h*>(p + 8 * h);
  const v8h hi = *reinterpret_cast<const v8h*>(p + 16 + 8 * h);
  v16h r;
#pragma unroll
  for (int j = 0; j < 8; ++j) { r[j] = lo[j]; r[8 + j] = hi[j]; }
  return r;
}

__device__ __forceinline__ v8f wmma_f32_16x16x32(v16h a, v16h b, v8f c) {
  return __builtin_amdgcn_wmma_f32_16x16x32_f16(false, a, false, b, (short)0, c,
                                                false, false);
}

// One K-chunk of a 32x32 output tile: 4 operand loads feed 4 independent WMMAs
// (2x data reuse per load, independent accumulators hide WMMA hazards).
__device__ __forceinline__ void gemm_step(const _Float16* ar0,
                                          const _Float16* ar1,
                                          const _Float16* wr0,
                                          const _Float16* wr1, int h,
                                          v8f acc[2][2]) {
  v16h a0 = load_op32(ar0, h);
  v16h a1 = load_op32(ar1, h);
  v16h b0 = load_op32(wr0, h);
  v16h b1 = load_op32(wr1, h);
  acc[0][0] = wmma_f32_16x16x32(a0, b0, acc[0][0]);
  acc[0][1] = wmma_f32_16x16x32(a0, b1, acc[0][1]);
  acc[1][0] = wmma_f32_16x16x32(a1, b0, acc[1][0]);
  acc[1][1] = wmma_f32_16x16x32(a1, b1, acc[1][1]);
}

// ---------------------------------------------------------------------------
// Utility kernels
// ---------------------------------------------------------------------------
__global__ void cvt_f32_to_f16_kernel(const float* __restrict__ in,
                                      _Float16* __restrict__ out, int n) {
  int i = blockIdx.x * blockDim.x + threadIdx.x;
  if (i < n) out[i] = (_Float16)in[i];
}

__global__ void zero_f32_kernel(float* __restrict__ p, int n) {
  int i = blockIdx.x * blockDim.x + threadIdx.x;
  if (i < n) p[i] = 0.0f;
}

// ---------------------------------------------------------------------------
// Phase feature kernel: feat[b][k][l][0..15]=cos(ph_p), [16..31]=sin(ph_p)
// ph = tanh(x_row . phase_w[k][p] + phase_b[k][p]) * pi
// One wave per (b,k,l); lane = p + 16*h; each half-lane sums 128 of 256 dims.
// ---------------------------------------------------------------------------
__global__ void feat_kernel(const float* __restrict__ xin,
                            const float* __restrict__ pw,
                            const float* __restrict__ pb,
                            _Float16* __restrict__ feat, int B, int K, int L,
                            int D) {
  int g = (blockIdx.x * blockDim.x + threadIdx.x) >> 5;
  if (g >= B * K * L) return;
  int lane = threadIdx.x & 31;
  int p = lane & 15, h = lane >> 4;
  int b = g / (K * L);
  int rem = g - b * (K * L);
  int k = rem / L;
  int l = rem - k * L;
  const float* xr = xin + ((size_t)(b * L + l)) * D + 128 * h;
  const float* wr = pw + ((size_t)(k * 16 + p)) * D + 128 * h;
  float s = 0.f;
#pragma unroll 8
  for (int d = 0; d < 128; ++d) s += xr[d] * wr[d];
  s += __shfl_xor(s, 16, 32);  // combine both halves -> full dot
  float ph = tanhf(s + pb[k * 16 + p]) * PI_F;
  float v = (h == 0) ? cosf(ph) : sinf(ph);
  feat[(((size_t)(b * K + k) * L + l)) * 32 + p + 16 * h] = (_Float16)v;
}

// ---------------------------------------------------------------------------
// LayerNorm over last dim N -> f16 output. One 256-thread block per row.
// ---------------------------------------------------------------------------
__global__ void ln_kernel(const float* __restrict__ in,
                          const float* __restrict__ w,
                          const float* __restrict__ bv,
                          _Float16* __restrict__ out, int N) {
  __shared__ float sred[256];
  int row = blockIdx.x;
  const float* rp = in + (size_t)row * N;
  float s = 0.f, s2 = 0.f;
  for (int j = threadIdx.x; j < N; j += 256) {
    float v = rp[j];
    s += v;
    s2 += v * v;
  }
  sred[threadIdx.x] = s;
  __syncthreads();
  for (int o = 128; o > 0; o >>= 1) {
    if (threadIdx.x < o) sred[threadIdx.x] += sred[threadIdx.x + o];
    __syncthreads();
  }
  float mean = sred[0] / (float)N;
  __syncthreads();
  sred[threadIdx.x] = s2;
  __syncthreads();
  for (int o = 128; o > 0; o >>= 1) {
    if (threadIdx.x < o) sred[threadIdx.x] += sred[threadIdx.x + o];
    __syncthreads();
  }
  float var = sred[0] / (float)N - mean * mean;
  float inv = rsqrtf(var + 1e-5f);
  for (int j = threadIdx.x; j < N; j += 256) {
    float v = (rp[j] - mean) * inv * w[j] + bv[j];
    out[(size_t)row * N + j] = (_Float16)v;
  }
}

// ---------------------------------------------------------------------------
// Generic WMMA GEMM:  C[m][n] = sum_kk A[m][kk] * W[n][kk] + bias[n]
// A split into A0 (first K0 cols) and A1 (remaining), row-major f16.
// Each wave computes one 32x32 tile (2x2 sub-tiles). Epilogues:
//   0 VT      : out16[(m/Lc)*N*Lc + n*Lc + (m%Lc)] = c         (V transpose)
//   1 F16     : out16[m*N+n] = c
//   2 GELU    : out16[m*N+n] = gelu_exact(c)
//   3 REFINED : fp0[m*N+n] += c ; out16[m*N+n] = c
//   4 GATE    : q = fp0[m*N+n] + tanh(c); fp0 = q; out16[m*N+n] = q
//   5 FINAL   : fp0[m*N+n] = fp1[m*N+n] + c
// ---------------------------------------------------------------------------
__global__ void gemm_wmma_kernel(const _Float16* __restrict__ A0,
                                 const _Float16* __restrict__ A1, int K0,
                                 int Kd, const _Float16* __restrict__ W,
                                 const float* __restrict__ bias, int M, int N,
                                 int ep, _Float16* __restrict__ out16,
                                 float* __restrict__ fp0,
                                 const float* __restrict__ fp1, int Lc) {
  int wid = (blockIdx.x * blockDim.x + threadIdx.x) >> 5;
  int nT = N >> 5;
  int mt = wid / nT, nt = wid - mt * nT;
  if (mt >= (M >> 5)) return;
  int lane = threadIdx.x & 31;
  int idx = lane & 15, h = lane >> 4;

  const _Float16* wr0 = W + (size_t)(nt * 32 + idx) * Kd;
  const _Float16* wr1 = W + (size_t)(nt * 32 + 16 + idx) * Kd;

  v8f acc[2][2];
#pragma unroll
  for (int i = 0; i < 2; ++i)
#pragma unroll
    for (int j = 0; j < 2; ++j) acc[i][j] = (v8f){};

  {  // A0 region: kk in [0, K0)
    const _Float16* ar0 = A0 + (size_t)(mt * 32 + idx) * K0;
    const _Float16* ar1 = A0 + (size_t)(mt * 32 + 16 + idx) * K0;
    for (int kk = 0; kk < K0; kk += 32)
      gemm_step(ar0 + kk, ar1 + kk, wr0 + kk, wr1 + kk, h, acc);
  }
  if (A1) {  // A1 region: kk in [K0, Kd)
    int K1 = Kd - K0;
    const _Float16* ar0 = A1 + (size_t)(mt * 32 + idx) * K1;
    const _Float16* ar1 = A1 + (size_t)(mt * 32 + 16 + idx) * K1;
    for (int kk = 0; kk < K1; kk += 32)
      gemm_step(ar0 + kk, ar1 + kk, wr0 + K0 + kk, wr1 + K0 + kk, h, acc);
  }

#pragma unroll
  for (int tj = 0; tj < 2; ++tj) {
    int n = nt * 32 + tj * 16 + idx;
    float bn = bias ? bias[n] : 0.f;
#pragma unroll
    for (int ti = 0; ti < 2; ++ti) {
#pragma unroll
      for (int r = 0; r < 8; ++r) {
        int m = mt * 32 + ti * 16 + r + 8 * h;
        float v = acc[ti][tj][r] + bn;
        size_t o = (size_t)m * N + n;
        switch (ep) {
          case 0: {  // V transpose: (B,L,N) -> (B,N,Lc)
            int bb = m / Lc, ll = m - bb * Lc;
            out16[((size_t)bb * N + n) * Lc + ll] = (_Float16)v;
          } break;
          case 1:
            out16[o] = (_Float16)v;
            break;
          case 2: {
            float g = 0.5f * v * (1.0f + erff(v * 0.70710678118654752f));
            out16[o] = (_Float16)g;
          } break;
          case 3:
            fp0[o] += v;
            out16[o] = (_Float16)v;
            break;
          case 4: {
            float q = fp0[o] + tanhf(v);
            fp0[o] = q;
            out16[o] = (_Float16)q;
          } break;
          case 5:
            fp0[o] = fp1[o] + v;
            break;
        }
      }
    }
  }
}

// ---------------------------------------------------------------------------
// Causal linear attention (no softmax):
//   combined[b][l][k*256 + d] = (1/(4*sqrt(l+1))) * sum_{l'<=l} (Q[l].K[l']) V[l',d]
// Q/K features: (B,K,L,32) f16; V transposed: (B,256,L) f16.
// Block = 4 waves; wave w owns 64-wide V slice; 16-row query tile per block.
// Streams keys in pairs of 16 (32 at a time -> WMMA K-dim).
// ---------------------------------------------------------------------------
__global__ void attn_kernel(const _Float16* __restrict__ Qf,
                            const _Float16* __restrict__ Kf,
                            const _Float16* __restrict__ Vt,
                            float* __restrict__ combined, int Lc) {
  __shared__ __align__(16) _Float16 slds[4 * 16 * 32];
  int wv = threadIdx.x >> 5;
  int lane = threadIdx.x & 31;
  int idx = lane & 15, h = lane >> 4;
  int qt = blockIdx.x, k = blockIdx.y, b = blockIdx.z;
  int ql0 = qt * 16;
  int nbase = wv * 64;

  const _Float16* qbase = Qf + (size_t)(b * 4 + k) * Lc * 32;
  const _Float16* kbase = Kf + (size_t)(b * 4 + k) * Lc * 32;
  const _Float16* vtb = Vt + (size_t)b * 256 * Lc;
  _Float16* sl = slds + wv * 512;

  v16h qa = load_op32(qbase + (size_t)(ql0 + idx) * 32, h);

  v8f acc[4];
#pragma unroll
  for (int cc = 0; cc < 4; ++cc) acc[cc] = (v8f){};

  int npair = (ql0 + 47) >> 5;  // 32-key pairs covering keys [0, ql0+15]
  for (int kp = 0; kp < npair; ++kp) {
    int kl0 = kp * 32;
    v16h b1 = load_op32(kbase + (size_t)(kl0 + idx) * 32, h);
    v16h b2 = load_op32(kbase + (size_t)(kl0 + 16 + idx) * 32, h);
    v8f s1 = {}, s2 = {};
    s1 = wmma_f32_16x16x32(qa, b1, s1);
    s2 = wmma_f32_16x16x32(qa, b2, s2);

    // causal mask + f32->f16, C-layout -> row-major 16x32 in per-wave LDS
#pragma unroll
    for (int r = 0; r < 8; ++r) {
      int Mr = r + 8 * h;
      int qg = ql0 + Mr;
      int c1 = kl0 + idx, c2 = kl0 + 16 + idx;
      sl[Mr * 32 + idx] = (_Float16)((c1 <= qg) ? s1[r] : 0.0f);
      sl[Mr * 32 + 16 + idx] = (_Float16)((c2 <= qg) ? s2[r] : 0.0f);
    }
    // re-read as A operand (per-wave region; same-wave DS ops are ordered)
    v16h sa = load_op32(sl + idx * 32, h);

#pragma unroll
    for (int cc = 0; cc < 4; ++cc) {
      v16h vb = load_op32(vtb + (size_t)(nbase + cc * 16 + idx) * Lc + kl0, h);
      acc[cc] = wmma_f32_16x16x32(sa, vb, acc[cc]);
    }
  }

#pragma unroll
  for (int r = 0; r < 8; ++r) {
    int Mr = r + 8 * h;
    int l = ql0 + Mr;
    float sc = rsqrtf((float)(l + 1)) * 0.25f;  // 1/(sqrt(l+1)*sqrt(P))
    size_t rowo = ((size_t)(b * Lc + l)) * 1024 + (size_t)k * 256;
#pragma unroll
    for (int cc = 0; cc < 4; ++cc)
      combined[rowo + nbase + cc * 16 + idx] = acc[cc][r] * sc;
  }
}

// ---------------------------------------------------------------------------
// Host side
// ---------------------------------------------------------------------------
extern "C" void kernel_launch(void* const* d_in, const int* in_sizes, int n_in,
                              void* d_out, int out_size, void* d_ws,
                              size_t ws_size, hipStream_t stream) {
  const int B = 2, L = 2048, D = 256, K = 4, I = 3;
  const int M = B * L;  // 4096 rows

  const float* x = (const float*)d_in[0];
  const float* phase_w = (const float*)d_in[1];
  const float* phase_b = (const float*)d_in[2];
  const float* val_w = (const float*)d_in[3];
  const float* val_b = (const float*)d_in[4];
  const float* ref_ln_w = (const float*)d_in[5];
  const float* ref_ln_b = (const float*)d_in[6];
  const float* ref1_w = (const float*)d_in[7];
  const float* ref1_b = (const float*)d_in[8];
  const float* ref2_w = (const float*)d_in[9];
  const float* ref2_b = (const float*)d_in[10];
  const float* qu_w = (const float*)d_in[11];
  const float* qu_b = (const float*)d_in[12];
  const float* out_ln_w = (const float*)d_in[13];
  const float* out_ln_b = (const float*)d_in[14];
  const float* out_w = (const float*)d_in[15];
  const float* out_b = (const float*)d_in[16];
  float* outp = (float*)d_out;

  char* wsb = (char*)d_ws;
  size_t off = 0;
  auto carve = [&](size_t bytes) -> char* {
    char* p = wsb + off;
    off = (off + bytes + 255) & ~(size_t)255;
    return p;
  };
  _Float16* x16 = (_Float16*)carve((size_t)M * D * 2);
  _Float16* valw16 = (_Float16*)carve((size_t)D * D * 2);
  _Float16* ref1w16 = (_Float16*)carve((size_t)I * 2 * D * 4 * D * 2);
  _Float16* ref2w16 = (_Float16*)carve((size_t)I * D * 2 * D * 2);
  _Float16* quw16 = (_Float16*)carve((size_t)I * D * 2 * D * 2);
  _Float16* outw16 = (_Float16*)carve((size_t)D * D * 2);
  _Float16* vt16 = (_Float16*)carve((size_t)B * D * L * 2);
  _Float16* kfeat = (_Float16*)carve((size_t)B * K * L * 32 * 2);
  _Float16* qfeat = (_Float16*)carve((size_t)B * K * L * 32 * 2);
  float* combined = (float*)carve((size_t)M * 4 * D * 4);
  _Float16* hln16 = (_Float16*)carve((size_t)M * 4 * D * 2);
  _Float16* h2_16 = (_Float16*)carve((size_t)M * 2 * D * 2);
  _Float16* ref16 = (_Float16*)carve((size_t)M * D * 2);
  float* accf = (float*)carve((size_t)M * D * 4);
  float* qf32 = (float*)carve((size_t)M * D * 4);
  _Float16* qf16a = (_Float16*)carve((size_t)M * D * 2);
  _Float16* qf16b = (_Float16*)carve((size_t)M * D * 2);
  _Float16* accln16 = (_Float16*)carve((size_t)M * D * 2);
  (void)ws_size;
  (void)n_in;
  (void)in_sizes;
  (void)out_size;

  auto cvt = [&](const float* in, _Float16* out, int n) {
    cvt_f32_to_f16_kernel<<<(n + 255) / 256, 256, 0, stream>>>(in, out, n);
  };
  auto gemm = [&](const _Float16* A0, const _Float16* A1, int K0, int Kd,
                  const _Float16* W, const float* bias, int Mm, int Nn, int ep,
                  _Float16* o16, float* f0, const float* f1, int Lc) {
    int waves = (Mm / 32) * (Nn / 32);
    gemm_wmma_kernel<<<(waves + 3) / 4, 128, 0, stream>>>(
        A0, A1, K0, Kd, W, bias, Mm, Nn, ep, o16, f0, f1, Lc);
  };

  // --- staging ---
  cvt(x, x16, M * D);
  cvt(val_w, valw16, D * D);
  cvt(ref1_w, ref1w16, I * 2 * D * 4 * D);
  cvt(ref2_w, ref2w16, I * D * 2 * D);
  cvt(qu_w, quw16, I * D * 2 * D);
  cvt(out_w, outw16, D * D);

  // K-side phase features (fixed across iterations)
  {
    int waves = B * K * L;
    feat_kernel<<<(waves * 32 + 127) / 128, 128, 0, stream>>>(
        x, phase_w, phase_b, kfeat, B, K, L, D);
  }
  // values = x @ val_w^T + val_b, stored transposed (B,D,L) f16
  gemm(x16, nullptr, D, D, valw16, val_b, M, D, /*VT*/ 0, vt16, nullptr,
       nullptr, L);

  // query = x ; accumulated = 0
  hipMemcpyAsync(qf32, x, (size_t)M * D * 4, hipMemcpyDeviceToDevice, stream);
  zero_f32_kernel<<<(M * D + 255) / 256, 256, 0, stream>>>(accf, M * D);

  _Float16* qcur16 = x16;
  _Float16* qnext16 = qf16a;
  for (int i = 0; i < I; ++i) {
    // Q-side phase features from current query
    int waves = B * K * L;
    feat_kernel<<<(waves * 32 + 127) / 128, 128, 0, stream>>>(
        qf32, phase_w, phase_b, qfeat, B, K, L, D);
    // causal linear attention -> combined (B,L,K*D)
    attn_kernel<<<dim3(L / 16, K, B), 128, 0, stream>>>(qfeat, kfeat, vt16,
                                                        combined, L);
    // LN over 4D
    ln_kernel<<<M, 256, 0, stream>>>(combined, ref_ln_w + (size_t)i * 4 * D,
                                     ref_ln_b + (size_t)i * 4 * D, hln16,
                                     4 * D);
    // h = gelu(hln @ ref1_w^T + b)
    gemm(hln16, nullptr, 4 * D, 4 * D, ref1w16 + (size_t)i * 2 * D * 4 * D,
         ref1_b + (size_t)i * 2 * D, M, 2 * D, /*GELU*/ 2, h2_16, nullptr,
         nullptr, 0);
    // refined = h @ ref2_w^T + b ; accumulated += refined
    gemm(h2_16, nullptr, 2 * D, 2 * D, ref2w16 + (size_t)i * D * 2 * D,
         ref2_b + (size_t)i * D, M, D, /*REFINED*/ 3, ref16, accf, nullptr, 0);
    // query += tanh([query, refined] @ qu_w^T + b)   (double-buffered f16)
    gemm(qcur16, ref16, D, 2 * D, quw16 + (size_t)i * D * 2 * D,
         qu_b + (size_t)i * D, M, D, /*GATE*/ 4, qnext16, qf32, nullptr, 0);
    _Float16* nn = (qnext16 == qf16a) ? qf16b : qf16a;
    qcur16 = qnext16;
    qnext16 = nn;
  }

  // out = x + LN(accumulated) @ out_w^T + out_b
  ln_kernel<<<M, 256, 0, stream>>>(accf, out_ln_w, out_ln_b, accln16, D);
  gemm(accln16, nullptr, D, D, outw16, out_b, M, D, /*FINAL*/ 5, nullptr, outp,
       x, 0);
}
